// HGTEncoder_18863496364559
// MI455X (gfx1250) — compile-verified, matrix operand order checked by reading, not run on hardware
//
#include <hip/hip_runtime.h>
#include <hip/hip_bf16.h>

typedef __attribute__((ext_vector_type(16))) _Float16 v16h;
typedef __attribute__((ext_vector_type(8)))  _Float16 v8h;
typedef __attribute__((ext_vector_type(8)))  float    v8f;

#define NHEAD 8
#define HDIM  16
#define HDTOT 128

// ---------------------------------------------------------------------------
// Utility kernels
// ---------------------------------------------------------------------------

__global__ void k_f32_to_f16(const float* __restrict__ in, _Float16* __restrict__ out, int n) {
  int i = blockIdx.x * blockDim.x + threadIdx.x;
  if (i < n) out[i] = (_Float16)in[i];
}

__global__ void k_zero_f32(float* __restrict__ p, int n) {
  int i = blockIdx.x * blockDim.x + threadIdx.x;
  if (i < n) p[i] = 0.0f;
}

__global__ void k_zero_u32(unsigned* __restrict__ p, int n) {
  int i = blockIdx.x * blockDim.x + threadIdx.x;
  if (i < n) p[i] = 0u;
}

// Fold per-head (16x16) relation transform into a 128x128 weight (or copy),
// converting to f16 in the WMMA B-fragment swizzled layout:
// for ktile(32 rows) x ntile(16 cols): lane l = half*16 + (n&15),
// element j holds K = ktile*32 + [half*8+j (j<8) | 16+half*8+(j-8)].
__global__ void k_pack_weight(const float* __restrict__ W, const float* __restrict__ rel,
                              _Float16* __restrict__ Bp) {
  int idx = blockIdx.x * blockDim.x + threadIdx.x;
  if (idx >= 128 * 128) return;
  int k = idx >> 7, n = idx & 127;
  float val;
  if (rel) {
    int h = n >> 4, e = n & 15;
    const float* wrow = W + k * 128 + h * 16;
    const float* ar   = rel + h * 256;   // [16][16]
    float s = 0.0f;
#pragma unroll
    for (int d0 = 0; d0 < 16; ++d0) s += wrow[d0] * ar[d0 * 16 + e];
    val = s;
  } else {
    val = W[idx];
  }
  int ktile = k >> 5, kk = k & 31;
  int half  = (kk >> 3) & 1;
  int j     = (kk & 7) + ((kk >> 4) << 3);
  int lane  = half * 16 + (n & 15);
  int ntile = n >> 4;
  Bp[(((ktile * 8 + ntile) * 32) + lane) * 16 + j] = (_Float16)val;
}

__global__ void k_fuse_bias(const float* __restrict__ b, const float* __restrict__ rel,
                            float* __restrict__ out) {
  int n = threadIdx.x;
  if (n >= 128) return;
  if (rel) {
    int h = n >> 4, e = n & 15;
    float s = 0.0f;
#pragma unroll
    for (int d0 = 0; d0 < 16; ++d0) s += b[h * 16 + d0] * rel[h * 256 + d0 * 16 + e];
    out[n] = s;
  } else {
    out[n] = b[n];
  }
}

// ---------------------------------------------------------------------------
// WMMA GEMM: out[M,128] = A_f16[M,128] x Wpacked[128,128] + bias
// Packed B (32 KB) is staged once per workgroup into LDS; all 8 waves read
// fragments via ds_load. mode 0: plain.
// mode 1: out = relu(beta*(acc+bias) + (1-beta)*xprev), beta=sigmoid(*skip)
// One wave per 16-row M tile; 8 N-tiles x 4 K-steps = 32 v_wmma per wave.
// ---------------------------------------------------------------------------
__global__ __launch_bounds__(256) void k_gemm128(
    const _Float16* __restrict__ A, const _Float16* __restrict__ Bp,
    const float* __restrict__ bias, float* __restrict__ out, int M,
    int mode, const float* __restrict__ skip, const float* __restrict__ xprev) {
  __shared__ _Float16 Bs[128 * 128];     // 32 KB packed weight tile

  // Cooperative stage: 256 threads x 8 x 16B = 32 KB
  {
    const uint4* gsrc = (const uint4*)Bp;
    uint4* ldst = (uint4*)Bs;
#pragma unroll
    for (int i = 0; i < 8; ++i) ldst[threadIdx.x + 256 * i] = gsrc[threadIdx.x + 256 * i];
  }
  __syncthreads();

  int wave = threadIdx.x >> 5;
  int lane = threadIdx.x & 31;
  int tileM = blockIdx.x * 8 + wave;
  bool active = (tileM * 16 < M);        // wave-uniform (EXEC all-1 inside)
  if (active) {
    int ml = lane & 15;
    int hh = lane >> 4;
    int row = tileM * 16 + ml;

    v8f acc[8];
#pragma unroll
    for (int i = 0; i < 8; ++i)
#pragma unroll
      for (int e = 0; e < 8; ++e) acc[i][e] = 0.0f;

#pragma unroll
    for (int kc = 0; kc < 4; ++kc) {
      v16h af;
      if (row < M) {
        v8h lo = *(const v8h*)(A + (size_t)row * 128 + kc * 32 + hh * 8);
        v8h hi = *(const v8h*)(A + (size_t)row * 128 + kc * 32 + 16 + hh * 8);
#pragma unroll
        for (int j = 0; j < 8; ++j) { af[j] = lo[j]; af[j + 8] = hi[j]; }
      } else {
#pragma unroll
        for (int j = 0; j < 16; ++j) af[j] = (_Float16)0.0f;
      }
#pragma unroll
      for (int nt = 0; nt < 8; ++nt) {
        v16h bf = *(const v16h*)(Bs + ((kc * 8 + nt) * 32 + lane) * 16);
        acc[nt] = __builtin_amdgcn_wmma_f32_16x16x32_f16(
            false, af, false, bf, (short)0, acc[nt], false, false);
      }
    }

    float beta = 1.0f, omb = 0.0f;
    if (mode == 1) {
      float s = *skip;
      beta = 1.0f / (1.0f + expf(-s));
      omb  = 1.0f - beta;
    }
    int rowbase = tileM * 16 + hh * 8;
#pragma unroll
    for (int nt = 0; nt < 8; ++nt) {
      int col = nt * 16 + ml;
      float bv = bias[col];
#pragma unroll
      for (int v = 0; v < 8; ++v) {
        int r = rowbase + v;
        if (r < M) {
          float val = acc[nt][v] + bv;
          if (mode == 1) {
            val = beta * val + omb * xprev[(size_t)r * 128 + col];
            val = fmaxf(val, 0.0f);
          }
          out[(size_t)r * 128 + col] = val;
        }
      }
    }
  }
}

// ---------------------------------------------------------------------------
// Edge-phase kernels (scatter softmax over incoming edges per dst node)
// ---------------------------------------------------------------------------
__device__ __forceinline__ unsigned fkey(float x) {
  unsigned u = __float_as_uint(x);
  return (u & 0x80000000u) ? ~u : (u | 0x80000000u);
}

__global__ void k_edge_logits(const int* __restrict__ src, const int* __restrict__ dst, int E,
                              const float* __restrict__ K, const float* __restrict__ Q,
                              const float* __restrict__ prel,
                              float* __restrict__ aout, unsigned* __restrict__ amaxk) {
  int t = blockIdx.x * blockDim.x + threadIdx.x;
  if (t >= E * 8) return;
  int e = t >> 3, h = t & 7;
  int s = src[e], d2 = dst[e];
  const float4* qp = (const float4*)(Q + (size_t)d2 * 128 + h * 16);
  const float4* kp = (const float4*)(K + (size_t)s  * 128 + h * 16);
  float a = 0.0f;
#pragma unroll
  for (int i = 0; i < 4; ++i) {
    float4 q = qp[i], k = kp[i];
    a += q.x * k.x + q.y * k.y + q.z * k.z + q.w * k.w;
  }
  a *= prel[h] * 0.25f;                 // 1/sqrt(16)
  aout[(size_t)e * 8 + h] = a;
  atomicMax(&amaxk[(size_t)d2 * 8 + h], fkey(a));
}

__global__ void k_finalize(const unsigned* __restrict__ amaxk, float* __restrict__ amaxf,
                           float* __restrict__ denom, int n) {
  int i = blockIdx.x * blockDim.x + threadIdx.x;
  if (i >= n) return;
  unsigned k = amaxk[i];
  unsigned u = (k & 0x80000000u) ? (k & 0x7FFFFFFFu) : ~k;
  float m = __uint_as_float(u);
  if (!(fabsf(m) < 3.0e38f)) m = 0.0f;   // non-finite (incl. untouched) -> 0
  amaxf[i] = m;
  denom[i] = 1e-16f;
}

__global__ void k_edge_exp(const int* __restrict__ dst, int E, float* __restrict__ a,
                           const float* __restrict__ amaxf, float* __restrict__ denom) {
  int t = blockIdx.x * blockDim.x + threadIdx.x;
  if (t >= E * 8) return;
  int e = t >> 3, h = t & 7;
  int d2 = dst[e];
  float ex = expf(a[(size_t)e * 8 + h] - amaxf[(size_t)d2 * 8 + h]);
  a[(size_t)e * 8 + h] = ex;
  atomicAdd(&denom[(size_t)d2 * 8 + h], ex);
}

__global__ void k_edge_agg(const int* __restrict__ src, const int* __restrict__ dst, int E,
                           const float* __restrict__ a, const float* __restrict__ denom,
                           const float* __restrict__ V, float* __restrict__ agg) {
  int t = blockIdx.x * blockDim.x + threadIdx.x;
  if (t >= E * 8) return;
  int e = t >> 3, h = t & 7;
  int s = src[e], d2 = dst[e];
  float alpha = a[(size_t)e * 8 + h] / denom[(size_t)d2 * 8 + h];
  const float* vp = V + (size_t)s * 128 + h * 16;
  float* ap = agg + (size_t)d2 * 128 + h * 16;
#pragma unroll
  for (int i = 0; i < 16; ++i) atomicAdd(ap + i, alpha * vp[i]);
}

__global__ void k_gelu_f16(const float* __restrict__ in, _Float16* __restrict__ out, int n) {
  int i = blockIdx.x * blockDim.x + threadIdx.x;
  if (i >= n) return;
  float x = in[i];
  float g = 0.5f * x * (1.0f + erff(x * 0.70710678118654752f));
  out[i] = (_Float16)g;
}

// ---------------------------------------------------------------------------
// Host side
// ---------------------------------------------------------------------------
extern "C" void kernel_launch(void* const* d_in, const int* in_sizes, int n_in,
                              void* d_out, int out_size, void* d_ws, size_t ws_size,
                              hipStream_t stream) {
  (void)in_sizes; (void)n_in; (void)out_size; (void)ws_size;

  // ---- problem constants ----
  static const int NTN[4]   = {20000, 30000, 5000, 200};
  static const int NTOFF[4] = {0, 20000, 50000, 55000};
  const int NTOT = 55200;
  static const int RE[8]   = {250000, 150000, 60000, 5000, 40000, 250000, 60000, 5000};
  static const int RSRC[8] = {0, 1, 1, 2, 2, 1, 2, 3};
  static const int RDST[8] = {1, 1, 2, 3, 2, 0, 1, 2};
  // per-destination-type concatenation offsets into the 820000-edge logit buffer
  static const long RAOFF[8] = {250000, 500000, 710000, 815000, 770000, 0, 650000, 810000};
  const int ETOT = 820000;

  // ---- inputs ----
  const float* x_in[4];
  for (int t = 0; t < 4; ++t) x_in[t] = (const float*)d_in[t];
  const int* ei[8];
  for (int r = 0; r < 8; ++r) ei[r] = (const int*)d_in[4 + r];
  // params: base 12; per layer 60 tensors: 4 node types x 9, then 8 relations x 3
  auto P = [&](int l, int idx) -> const float* { return (const float*)d_in[12 + 60 * l + idx]; };

  // ---- workspace carve-up ----
  char* ws = (char*)d_ws;
  const size_t OFF_XF16  = 0;                                        // 55200*128 f16
  const size_t OFF_QAGG  = OFF_XF16  + (size_t)NTOT * 128 * 2;       // 55200*128 f32 (Q, then agg)
  const size_t OFF_KV    = OFF_QAGG  + (size_t)NTOT * 128 * 4;       // 30000*128 f32 (K/V per relation)
  const size_t OFF_ABUF  = OFF_KV    + (size_t)30000 * 128 * 4;      // 820000*8 f32
  const size_t OFF_AMAXK = OFF_ABUF  + (size_t)ETOT * 8 * 4;         // 55200*8 u32
  const size_t OFF_AMAXF = OFF_AMAXK + (size_t)NTOT * 8 * 4;         // 55200*8 f32
  const size_t OFF_DEN   = OFF_AMAXF + (size_t)NTOT * 8 * 4;         // 55200*8 f32
  const size_t OFF_XNXT  = OFF_DEN   + (size_t)NTOT * 8 * 4;         // 55200*128 f32
  const size_t OFF_WP    = OFF_XNXT  + (size_t)NTOT * 128 * 4;       // 128*128 f16 packed weight
  const size_t OFF_BB    = OFF_WP + 32768;                           // 128 f32 bias

  _Float16* xf16  = (_Float16*)(ws + OFF_XF16);
  float*    qagg  = (float*)   (ws + OFF_QAGG);
  float*    kv    = (float*)   (ws + OFF_KV);
  float*    abuf  = (float*)   (ws + OFF_ABUF);
  unsigned* amaxk = (unsigned*)(ws + OFF_AMAXK);
  float*    amaxf = (float*)   (ws + OFF_AMAXF);
  float*    denom = (float*)   (ws + OFF_DEN);
  float*    xnext = (float*)   (ws + OFF_XNXT);
  _Float16* wpack = (_Float16*)(ws + OFF_WP);
  float*    bbuf  = (float*)   (ws + OFF_BB);
  float*    outf  = (float*)d_out;

  auto gemm_blocks = [](int M) { int tiles = (M + 15) / 16; return (tiles + 7) / 8; };

  for (int l = 0; l < 2; ++l) {
    // layer inputs
    const float* xl[4];
    for (int t = 0; t < 4; ++t)
      xl[t] = (l == 0) ? x_in[t] : (xnext + (size_t)NTOFF[t] * 128);

    // 1) x -> f16
    for (int t = 0; t < 4; ++t) {
      int n = NTN[t] * 128;
      k_f32_to_f16<<<(n + 255) / 256, 256, 0, stream>>>(xl[t], xf16 + (size_t)NTOFF[t] * 128, n);
    }

    // 2) Q projections per node type (WMMA GEMM)
    for (int t = 0; t < 4; ++t) {
      k_pack_weight<<<64, 256, 0, stream>>>(P(l, 9 * t + 2), nullptr, wpack);
      k_fuse_bias<<<1, 128, 0, stream>>>(P(l, 9 * t + 3), nullptr, bbuf);
      k_gemm128<<<gemm_blocks(NTN[t]), 256, 0, stream>>>(
          xf16 + (size_t)NTOFF[t] * 128, wpack, bbuf,
          qagg + (size_t)NTOFF[t] * 128, NTN[t], 0, nullptr, nullptr);
    }

    // 3) attention logits per relation (fused K GEMM + edge dot + atomic max)
    k_zero_u32<<<(NTOT * 8 + 255) / 256, 256, 0, stream>>>(amaxk, NTOT * 8);
    for (int r = 0; r < 8; ++r) {
      int s = RSRC[r], t = RDST[r];
      k_pack_weight<<<64, 256, 0, stream>>>(P(l, 9 * s + 0), P(l, 36 + 3 * r + 0), wpack);
      k_fuse_bias<<<1, 128, 0, stream>>>(P(l, 9 * s + 1), P(l, 36 + 3 * r + 0), bbuf);
      k_gemm128<<<gemm_blocks(NTN[s]), 256, 0, stream>>>(
          xf16 + (size_t)NTOFF[s] * 128, wpack, bbuf, kv, NTN[s], 0, nullptr, nullptr);
      int nth = RE[r] * 8;
      k_edge_logits<<<(nth + 255) / 256, 256, 0, stream>>>(
          ei[r], ei[r] + RE[r], RE[r], kv,
          qagg + (size_t)NTOFF[t] * 128, P(l, 36 + 3 * r + 2),
          abuf + (size_t)RAOFF[r] * 8, amaxk + (size_t)NTOFF[t] * 8);
    }

    // 4) finalize max, init denom; 5) exp + denom accumulation
    k_finalize<<<(NTOT * 8 + 255) / 256, 256, 0, stream>>>(amaxk, amaxf, denom, NTOT * 8);
    for (int r = 0; r < 8; ++r) {
      int t = RDST[r];
      int nth = RE[r] * 8;
      k_edge_exp<<<(nth + 255) / 256, 256, 0, stream>>>(
          ei[r] + RE[r], RE[r], abuf + (size_t)RAOFF[r] * 8,
          amaxf + (size_t)NTOFF[t] * 8, denom + (size_t)NTOFF[t] * 8);
    }

    // 6) aggregate: zero agg (reuses Q buffer), V GEMM per relation, scatter-add
    k_zero_f32<<<(NTOT * 128 + 255) / 256, 256, 0, stream>>>(qagg, NTOT * 128);
    for (int r = 0; r < 8; ++r) {
      int s = RSRC[r], t = RDST[r];
      k_pack_weight<<<64, 256, 0, stream>>>(P(l, 9 * s + 4), P(l, 36 + 3 * r + 1), wpack);
      k_fuse_bias<<<1, 128, 0, stream>>>(P(l, 9 * s + 5), P(l, 36 + 3 * r + 1), bbuf);
      k_gemm128<<<gemm_blocks(NTN[s]), 256, 0, stream>>>(
          xf16 + (size_t)NTOFF[s] * 128, wpack, bbuf, kv, NTN[s], 0, nullptr, nullptr);
      int nth = RE[r] * 8;
      k_edge_agg<<<(nth + 255) / 256, 256, 0, stream>>>(
          ei[r], ei[r] + RE[r], RE[r], abuf + (size_t)RAOFF[r] * 8,
          denom + (size_t)NTOFF[t] * 8, kv, qagg + (size_t)NTOFF[t] * 128);
    }

    // 7) gelu(agg) -> f16 (reuse xf16), then Wa GEMM with skip+relu epilogue
    k_gelu_f16<<<(NTOT * 128 + 255) / 256, 256, 0, stream>>>(qagg, xf16, NTOT * 128);
    for (int t = 0; t < 4; ++t) {
      k_pack_weight<<<64, 256, 0, stream>>>(P(l, 9 * t + 6), nullptr, wpack);
      k_fuse_bias<<<1, 128, 0, stream>>>(P(l, 9 * t + 7), nullptr, bbuf);
      float* outp = (l == 0) ? (xnext + (size_t)NTOFF[t] * 128)
                             : (outf + (size_t)NTOFF[t] * 128);
      k_gemm128<<<gemm_blocks(NTN[t]), 256, 0, stream>>>(
          xf16 + (size_t)NTOFF[t] * 128, wpack, bbuf, outp, NTN[t],
          1, P(l, 9 * t + 8), xl[t]);
    }
  }
}
